// BiMPM_58695023067402
// MI455X (gfx1250) — matrix operand order repaired and checked by reading, not compiled, not run on hardware
//
#include <hip/hip_runtime.h>

typedef __attribute__((ext_vector_type(16))) _Float16 v16h;
typedef __attribute__((ext_vector_type(8)))  float    v8f;

#define B_   32
#define L_   128
#define E_   300
#define H_   100
#define P_   20
#define D_   200   // 2H
#define G4H  400   // 4H
#define KP_  224   // D padded to 7*32
#define BLP  (B_ * L_ * P_)

// ---------------- monotonic float<->uint mapping for LDS uint atomicMax -----
__device__ __forceinline__ unsigned fmap(float f) {
    unsigned b = __float_as_uint(f);
    return b ^ ((unsigned)((int)b >> 31) | 0x80000000u);
}
__device__ __forceinline__ float funmap(unsigned u) {
    unsigned b = u ^ ((u & 0x80000000u) ? 0x80000000u : 0xFFFFFFFFu);
    return __uint_as_float(b);
}
__device__ __forceinline__ float sigmoidf_(float x) { return 1.f / (1.f + __expf(-x)); }

// ---------------- WMMA fragment builders ------------------------------------
// A-matrix 16x32 f16 layout (ISA 7.12.2): lanes 0-15 row M=lane, K 0..7 in
// v0..3 and K 16..23 in v4..7; lanes 16-31 row M=lane-16, K 8..15 / 24..31.
__device__ __forceinline__ v16h fragA_global_f32(const float* rowp, int kbase,
                                                 int K, int lane) {
    int khalf = (lane & 16) ? 8 : 0;
    v16h f;
#pragma unroll
    for (int e = 0; e < 8; ++e) {
        int k1 = kbase + khalf + e;
        int k2 = kbase + 16 + khalf + e;
        f[e]     = (k1 < K) ? (_Float16)rowp[k1] : (_Float16)0.f;
        f[e + 8] = (k2 < K) ? (_Float16)rowp[k2] : (_Float16)0.f;
    }
    return f;
}
__device__ __forceinline__ v16h fragA_lds(const _Float16* rowp, int kbase, int lane) {
    int khalf = (lane & 16) ? 8 : 0;
    v16h f;
#pragma unroll
    for (int e = 0; e < 8; ++e) {
        f[e]     = rowp[kbase + khalf + e];
        f[e + 8] = rowp[kbase + 16 + khalf + e];
    }
    return f;
}
// B-matrix 32x16 f16: lane holds column N=lane%16; lanes 0-15 carry K 0..7 and
// 16..23, lanes 16-31 carry K 8..15 and 24..31 (same K order as A).
// Source is row-major [N][K] (we compute C = U * V^T).
__device__ __forceinline__ v16h fragB_global_f32(const float* colrowp, int kbase,
                                                 int K, int lane) {
    int ko = (lane & 16) ? 8 : 0;
    v16h f;
#pragma unroll
    for (int e = 0; e < 16; ++e) {
        int k = kbase + ko + (e < 8 ? e : e + 8);
        f[e] = (k < K) ? (_Float16)colrowp[k] : (_Float16)0.f;
    }
    return f;
}
__device__ __forceinline__ v16h fragB_lds(const _Float16* colrowp, int kbase, int lane) {
    int ko = (lane & 16) ? 8 : 0;
    v16h f;
#pragma unroll
    for (int e = 0; e < 16; ++e) f[e] = colrowp[kbase + ko + (e < 8 ? e : e + 8)];
    return f;
}

// ---------------- Stage A: embedding gather ---------------------------------
__global__ void gather_embed(const int* __restrict__ s1, const int* __restrict__ s2,
                             const float* __restrict__ embed,
                             float* __restrict__ e1, float* __restrict__ e2) {
    int tid = blockIdx.x * blockDim.x + threadIdx.x;
    const int total = B_ * L_ * E_;
    if (tid >= total) return;
    int d  = tid % E_;
    int bt = tid / E_;
    e1[tid] = embed[(size_t)s1[bt] * E_ + d];
    e2[tid] = embed[(size_t)s2[bt] * E_ + d];
}

// ---------------- Stage B: input projection GEMM (WMMA f16) -----------------
// G[m][n] = sum_k E[m][k] * Wih[n][k] + bih[n] + bhh[n];  M=4096, N=400, K=300
__global__ void gemm_inproj(const float* __restrict__ Esrc,
                            const float* __restrict__ Wih,
                            const float* __restrict__ bih,
                            const float* __restrict__ bhh,
                            float* __restrict__ Gpre) {
    const int NT = G4H / 16;               // 25 tiles in N
    const int MT = (B_ * L_) / 16;         // 256 tiles in M
    int wave = (blockIdx.x * blockDim.x + threadIdx.x) >> 5;
    int lane = threadIdx.x & 31;
    if (wave >= MT * NT) return;
    int mt = wave / NT, nt = wave % NT;
    int m0 = mt * 16, n0 = nt * 16;
    const float* arow = Esrc + (size_t)(m0 + (lane & 15)) * E_;
    const float* brow = Wih  + (size_t)(n0 + (lane & 15)) * E_;
    v8f acc = {};
#pragma unroll
    for (int ks = 0; ks < 10; ++ks) {      // K=300 padded to 320
        v16h a = fragA_global_f32(arow, ks * 32, E_, lane);
        v16h b = fragB_global_f32(brow, ks * 32, E_, lane);
        acc = __builtin_amdgcn_wmma_f32_16x16x32_f16(false, a, false, b,
                                                     (short)0, acc, false, false);
    }
    int n = n0 + (lane & 15);
    float bias = bih[n] + bhh[n];
    int mb = (lane & 16) ? 8 : 0;
#pragma unroll
    for (int r = 0; r < 8; ++r) {
        int m = m0 + r + mb;
        Gpre[(size_t)m * G4H + n] = acc[r] + bias;
    }
}

// ---------------- Stage C: sequential LSTM recurrence (4 chains) ------------
// chain: 0=l11 fwd, 1=l11 bwd, 2=l12 fwd, 3=l12 bwd
__global__ void lstm_rec(const float* __restrict__ whh0, const float* __restrict__ whh1,
                         const float* __restrict__ whh2, const float* __restrict__ whh3,
                         const float* __restrict__ gbase,
                         float* __restrict__ s1e, float* __restrict__ s2e) {
    int chain = blockIdx.x;
    const float* whh = (chain == 0) ? whh0 : (chain == 1) ? whh1
                     : (chain == 2) ? whh2 : whh3;
    const float* G   = gbase + (size_t)chain * (B_ * L_ * G4H);
    float* out       = (chain < 2) ? s1e : s2e;
    const int off    = (chain & 1) ? H_ : 0;
    const bool rev   = (chain & 1);

    __shared__ float hS[B_][H_];
    __shared__ float cS[B_][H_];
    __shared__ float gS[B_][G4H];
    int tid = threadIdx.x;
    for (int i = tid; i < B_ * H_; i += blockDim.x) {
        hS[i / H_][i % H_] = 0.f;
        cS[i / H_][i % H_] = 0.f;
    }
    __syncthreads();

    for (int s = 0; s < L_; ++s) {
        int t = rev ? (L_ - 1 - s) : s;
        // g = Gpre[:, t, :] + h @ Whh^T
        for (int idx = tid; idx < B_ * G4H; idx += blockDim.x) {
            int bb = idx / G4H, n = idx % G4H;
            float acc = G[((size_t)bb * L_ + t) * G4H + n];
            const float* wr = whh + (size_t)n * H_;
            const float* hr = &hS[bb][0];
#pragma unroll 4
            for (int k = 0; k < H_; ++k) acc += hr[k] * wr[k];
            gS[bb][n] = acc;
        }
        __syncthreads();
        // gates (i, f, g, o)
        for (int idx = tid; idx < B_ * H_; idx += blockDim.x) {
            int bb = idx / H_, j = idx % H_;
            float ig = gS[bb][j];
            float fg = gS[bb][j + H_];
            float gg = gS[bb][j + 2 * H_];
            float og = gS[bb][j + 3 * H_];
            float cc = sigmoidf_(fg) * cS[bb][j] + sigmoidf_(ig) * tanhf(gg);
            float hh = sigmoidf_(og) * tanhf(cc);
            cS[bb][j] = cc;
            hS[bb][j] = hh;
            out[((size_t)bb * L_ + t) * D_ + off + j] = hh;
        }
        __syncthreads();
    }
}

// ---------------- Stage D: multi-perspective cosine + max (WMMA f16) --------
// One block per (b, p, w). Builds normalized U (s1*Wp) and V (s2*Wp) in LDS as
// f16, computes C = U_n * V_n^T (128x128) with WMMA, reduces row/col maxes.
__global__ void mpcos_match(const float* __restrict__ s1e, const float* __restrict__ s2e,
                            const float* __restrict__ W1, const float* __restrict__ W2,
                            float* __restrict__ out) {
    int b = blockIdx.x, p = blockIdx.y, w = blockIdx.z;
    const float* W = ((w == 0) ? W1 : W2) + (size_t)p * D_;

    __shared__ _Float16 U[L_][KP_];
    __shared__ _Float16 V[L_][KP_];
    __shared__ unsigned rowMax[L_];
    __shared__ unsigned colMax[L_];

    int tid  = threadIdx.x;
    int lane = tid & 31;
    int wv   = tid >> 5;                // 8 waves
    if (tid < L_) { rowMax[tid] = 0u; colMax[tid] = 0u; }

    // Build normalized f16 rows (one wave per row, round-robin).
    for (int side = 0; side < 2; ++side) {
        const float* S = side ? s2e : s1e;
        _Float16 (*Dst)[KP_] = side ? V : U;
        for (int i = wv; i < L_; i += 8) {
            const float* row = S + ((size_t)b * L_ + i) * D_;
            float ss = 0.f;
            for (int d = lane; d < D_; d += 32) {
                float v = row[d] * W[d];
                ss += v * v;
            }
#pragma unroll
            for (int o = 16; o; o >>= 1) ss += __shfl_xor(ss, o, 32);
            float inv = 1.f / fmaxf(sqrtf(ss), 1e-8f);
            for (int d = lane; d < KP_; d += 32)
                Dst[i][d] = (d < D_) ? (_Float16)(row[d] * W[d] * inv) : (_Float16)0.f;
        }
    }
    __syncthreads();

    // 8x8 grid of 16x16 output tiles; each wave owns 8 tiles.
    for (int tile = wv; tile < 64; tile += 8) {
        int ti = tile >> 3, tj = tile & 7;
        const _Float16* arow = &U[ti * 16 + (lane & 15)][0];
        const _Float16* brow = &V[tj * 16 + (lane & 15)][0];
        v8f acc = {};
#pragma unroll
        for (int ks = 0; ks < KP_ / 32; ++ks) {
            v16h a = fragA_lds(arow, ks * 32, lane);
            v16h bb = fragB_lds(brow, ks * 32, lane);
            acc = __builtin_amdgcn_wmma_f32_16x16x32_f16(false, a, false, bb,
                                                         (short)0, acc, false, false);
        }
        int n  = tj * 16 + (lane & 15);
        int mb = (lane & 16) ? 8 : 0;
#pragma unroll
        for (int r = 0; r < 8; ++r) {
            int m = ti * 16 + r + mb;
            unsigned mv = fmap(acc[r]);
            atomicMax(&rowMax[m], mv);   // max over j -> s1 match
            atomicMax(&colMax[n], mv);   // max over i -> s2 match
        }
    }
    __syncthreads();

    if (tid < L_) {
        float r1 = funmap(rowMax[tid]);
        float r2 = funmap(colMax[tid]);
        // out = concat(s1f, s1b, s2f, s2b), each [B, L, P]
        out[(size_t)w * BLP + ((size_t)b * L_ + tid) * P_ + p]       = r1;
        out[(size_t)(2 + w) * BLP + ((size_t)b * L_ + tid) * P_ + p] = r2;
    }
}

// ---------------- launcher ---------------------------------------------------
extern "C" void kernel_launch(void* const* d_in, const int* in_sizes, int n_in,
                              void* d_out, int out_size, void* d_ws, size_t ws_size,
                              hipStream_t stream) {
    const int*   s1    = (const int*)d_in[0];
    const int*   s2    = (const int*)d_in[1];
    const float* embed = (const float*)d_in[2];
    const float* l11_wih_f = (const float*)d_in[3];
    const float* l11_whh_f = (const float*)d_in[4];
    const float* l11_bih_f = (const float*)d_in[5];
    const float* l11_bhh_f = (const float*)d_in[6];
    const float* l11_wih_b = (const float*)d_in[7];
    const float* l11_whh_b = (const float*)d_in[8];
    const float* l11_bih_b = (const float*)d_in[9];
    const float* l11_bhh_b = (const float*)d_in[10];
    const float* l12_wih_f = (const float*)d_in[11];
    const float* l12_whh_f = (const float*)d_in[12];
    const float* l12_bih_f = (const float*)d_in[13];
    const float* l12_bhh_f = (const float*)d_in[14];
    const float* l12_wih_b = (const float*)d_in[15];
    const float* l12_whh_b = (const float*)d_in[16];
    const float* l12_bih_b = (const float*)d_in[17];
    const float* l12_bhh_b = (const float*)d_in[18];
    const float* W1 = (const float*)d_in[19];
    const float* W2 = (const float*)d_in[20];

    float* ws  = (float*)d_ws;
    const size_t nE = (size_t)B_ * L_ * E_;   // 1,228,800
    const size_t nG = (size_t)B_ * L_ * G4H;  // 1,638,400
    const size_t nS = (size_t)B_ * L_ * D_;   //   819,200
    float* e1  = ws;
    float* e2  = ws + nE;
    float* g   = ws + 2 * nE;
    float* s1e = ws + 2 * nE + 4 * nG;
    float* s2e = ws + 2 * nE + 4 * nG + nS;

    // A: gather
    {
        int total = B_ * L_ * E_;
        gather_embed<<<(total + 255) / 256, 256, 0, stream>>>(s1, s2, embed, e1, e2);
    }
    // B: four input projections (WMMA)
    {
        int waves  = ((B_ * L_) / 16) * (G4H / 16);   // 6400
        int blocks = waves / 8;                        // 800
        gemm_inproj<<<blocks, 256, 0, stream>>>(e1, l11_wih_f, l11_bih_f, l11_bhh_f, g);
        gemm_inproj<<<blocks, 256, 0, stream>>>(e1, l11_wih_b, l11_bih_b, l11_bhh_b, g + nG);
        gemm_inproj<<<blocks, 256, 0, stream>>>(e2, l12_wih_f, l12_bih_f, l12_bhh_f, g + 2 * nG);
        gemm_inproj<<<blocks, 256, 0, stream>>>(e2, l12_wih_b, l12_bih_b, l12_bhh_b, g + 3 * nG);
    }
    // C: recurrences (4 independent chains)
    lstm_rec<<<4, 256, 0, stream>>>(l11_whh_f, l11_whh_b, l12_whh_f, l12_whh_b, g, s1e, s2e);
    // D: cosine match (WMMA) — one block per (b, p, {W1,W2})
    mpcos_match<<<dim3(B_, P_, 2), 256, 0, stream>>>(s1e, s2e, W1, W2, (float*)d_out);
}
